// FlexAttnProcessor_46806553592110
// MI455X (gfx1250) — matrix operand order repaired and checked by reading, not compiled, not run on hardware
//
#include <hip/hip_runtime.h>
#include <hip/hip_bf16.h>

// ---------------------------------------------------------------------------
// FlexAttn processor for MI455X (gfx1250), bf16 WMMA path, round 4.
//   B=1, S_IMG=3840, S_TXT=256 (S=4096), C=1024, H=16, D=64
// Pipeline:
//   k0a: x = concat(enc,hid) fp32 -> bf16 [s][c]          (workspace)
//   k0b: Wq/Wk/Wv fp32 [K][N] -> bf16 [N][K]              (workspace)
//   k1 : barrier-free QKV GEMM (16x64/wave), ping-pong double-buffered
//        fragments (no register rotation), + bias + RoPE + q*=1/8
//        q,k -> bf16 [h][s][d];  v -> bf16 [h][d][s] (pre-transposed)
//   k2 : flash attention, 64-key chunks, double-buffered K/V LDS staging,
//        one barrier per chunk, 16 WMMA/chunk, fp32 out [s][h*d]
// ---------------------------------------------------------------------------

typedef __attribute__((ext_vector_type(16))) __bf16 bf16x16;
typedef __attribute__((ext_vector_type(8)))  float  f32x8;

static constexpr int S_TOT = 4096;
static constexpr int S_TXT = 256;
static constexpr int C_DIM = 1024;
static constexpr int H_N   = 16;
static constexpr int D_H   = 64;

union FragB16 { bf16x16 v; uint4 q[2]; __bf16 h[16]; };

static __device__ __forceinline__ void wait_dscnt0() {
#if __has_builtin(__builtin_amdgcn_s_wait_dscnt)
  __builtin_amdgcn_s_wait_dscnt(0);
#else
  asm volatile("s_wait_dscnt 0" ::: "memory");
#endif
}

// ---------------------------------------------------------------------------
// Kernel 0a: x = concat(encoder, hidden) -> bf16, row-major [4096][1024].
// ---------------------------------------------------------------------------
__global__ void __launch_bounds__(256)
xcat_bf16_kernel(const float* __restrict__ hid, const float* __restrict__ enc,
                 __bf16* __restrict__ xbf) {
  const size_t i = ((size_t)blockIdx.x * 256 + threadIdx.x) * 8;
  const int s = (int)(i >> 10);
  const int c = (int)(i & 1023);
  const float* src = (s < S_TXT) ? (enc + (size_t)s * C_DIM + c)
                                 : (hid + (size_t)(s - S_TXT) * C_DIM + c);
  const float4 a = reinterpret_cast<const float4*>(src)[0];
  const float4 b = reinterpret_cast<const float4*>(src)[1];
  union { uint4 q; __bf16 h[8]; } o;
  o.h[0] = (__bf16)a.x; o.h[1] = (__bf16)a.y; o.h[2] = (__bf16)a.z; o.h[3] = (__bf16)a.w;
  o.h[4] = (__bf16)b.x; o.h[5] = (__bf16)b.y; o.h[6] = (__bf16)b.z; o.h[7] = (__bf16)b.w;
  *reinterpret_cast<uint4*>(xbf + i) = o.q;
}

// ---------------------------------------------------------------------------
// Kernel 0b: W fp32 [K][N] -> wt bf16 [N][K] (B-fragment K-runs contiguous).
// ---------------------------------------------------------------------------
__global__ void wt_transpose_kernel(const float* __restrict__ Wq,
                                    const float* __restrict__ Wk,
                                    const float* __restrict__ Wv,
                                    __bf16* __restrict__ wt) {
  const int z = blockIdx.z;
  const float* W = (z == 0) ? Wq : ((z == 1) ? Wk : Wv);
  const int n = blockIdx.x * 16 + threadIdx.x;
  const int k = blockIdx.y * 16 + threadIdx.y;
  wt[(size_t)z * C_DIM * C_DIM + (size_t)n * C_DIM + k] =
      (__bf16)W[(size_t)k * C_DIM + n];
}

// ---------------------------------------------------------------------------
// Kernel 1: QKV projection, 16x64 per wave, ping-pong pipelined k-loop
// (two explicit fragment sets, no rotation copies).
// Grid: (S/16/4, C/64, 3), block = 128 (4 independent waves, no LDS/barriers).
// ---------------------------------------------------------------------------
__global__ void __launch_bounds__(128)
qkv_rope_kernel(const __bf16* __restrict__ xbf, const __bf16* __restrict__ wt,
                const float* __restrict__ rope_cos, const float* __restrict__ rope_sin,
                const float* __restrict__ bq, const float* __restrict__ bk,
                const float* __restrict__ bv,
                __bf16* __restrict__ qb, __bf16* __restrict__ kb,
                __bf16* __restrict__ vt) {
  const int lane = threadIdx.x & 31;
  const int wave = threadIdx.x >> 5;
  const int g    = lane >> 4;            // lane half (ISA fragment layout)
  const int ln   = lane & 15;
  const int m0   = (blockIdx.x * 4 + wave) * 16;   // 16 rows of x
  const int n0   = blockIdx.y * 64;                // 64 output channels
  const int z    = blockIdx.z;                     // 0=q 1=k 2=v
  const __bf16* wtz  = wt + (size_t)z * C_DIM * C_DIM;
  const __bf16* xrow = xbf + (size_t)(m0 + ln) * C_DIM;

  f32x8 acc[4];
#pragma unroll
  for (int t = 0; t < 4; ++t) acc[t] = (f32x8){0.f,0.f,0.f,0.f,0.f,0.f,0.f,0.f};

  auto loadA = [&](int kk, FragB16& a) {
    a.q[0] = *reinterpret_cast<const uint4*>(xrow + kk + g * 8);
    a.q[1] = *reinterpret_cast<const uint4*>(xrow + kk + 16 + g * 8);
  };
  auto loadB = [&](int kk, FragB16* b) {
#pragma unroll
    for (int t = 0; t < 4; ++t) {
      const __bf16* base = wtz + (size_t)(n0 + t * 16 + ln) * C_DIM + kk + g * 8;
      b[t].q[0] = reinterpret_cast<const uint4*>(base)[0];
      b[t].q[1] = reinterpret_cast<const uint4*>(base + 16)[0];
    }
  };
  auto mma4 = [&](const FragB16& a, const FragB16* b) {
#pragma unroll
    for (int t = 0; t < 4; ++t)
      acc[t] = __builtin_amdgcn_wmma_f32_16x16x32_bf16(
          false, a.v, false, b[t].v, (short)0, acc[t], false, false);
  };

  // ping-pong: set A computes while set B loads, then roles swap (no copies)
  FragB16 aA, bA[4], aB, bB[4];
  loadA(0, aA);
  loadB(0, bA);
  for (int kk = 0; kk < C_DIM; kk += 64) {
    loadA(kk + 32, aB);                      // in flight during mma on set A
    loadB(kk + 32, bB);
    mma4(aA, bA);
    if (kk + 64 < C_DIM) {
      loadA(kk + 64, aA);                    // in flight during mma on set B
      loadB(kk + 64, bA);
    }
    mma4(aB, bB);
  }

  // ---- epilogue: bias, RoPE (image rows of q/k), scale q, store bf16
  const float* bias = (z == 0) ? bq : ((z == 1) ? bk : bv);
#pragma unroll
  for (int t = 0; t < 4; ++t) {
    const int n = n0 + t * 16 + ln;
    const float bn = bias[n];
    const int d = n & 63;
    const int h = n >> 6;
#pragma unroll
    for (int r = 0; r < 8; ++r) {
      float c = acc[t][r] + bn;
      const int srow = m0 + r + 8 * g;         // C layout: M = r + 8*half
      if (z < 2 && srow >= S_TXT) {            // uniform across the tile
        const float cosv = rope_cos[(size_t)(srow - S_TXT) * D_H + d];
        const float sinv = rope_sin[(size_t)(srow - S_TXT) * D_H + d];
        const float partner = __shfl_xor(c, 1, 32);   // pair lane d^1
        const float rot = (d & 1) ? partner : -partner;
        c = c * cosv + rot * sinv;
      }
      if (z == 0) c *= 0.125f;                 // 1/sqrt(64) folded into q
      if (z == 2) {                            // V stored transposed [h][d][s]
        vt[((size_t)h * D_H + d) * S_TOT + srow] = (__bf16)c;
      } else {
        __bf16* dst = (z == 0) ? qb : kb;      // [h][s][d]
        dst[((size_t)h * S_TOT + srow) * D_H + d] = (__bf16)c;
      }
    }
  }
}

// ---------------------------------------------------------------------------
// Kernel 2: flash attention. WG = (head, 4x16 query rows). 64-key chunks,
// double-buffered LDS for K [key][d] and V^T [d][key]; next chunk's global
// loads are issued before the barrier and overlap the whole compute phase.
// One barrier per chunk; 16 WMMA per chunk. Grid: (S/64, H), block = 128.
// ---------------------------------------------------------------------------
__global__ void __launch_bounds__(128)
attn_kernel(const __bf16* __restrict__ qb, const __bf16* __restrict__ kbuf,
            const __bf16* __restrict__ vt, float* __restrict__ out) {
  __shared__ __align__(16) __bf16 lsK[2][64 * 64];  // [buf][key][d]  (16 KB)
  __shared__ __align__(16) __bf16 lsV[2][64 * 64];  // [buf][d][key]  (16 KB)
  __shared__ __align__(16) __bf16 lsP[4][16 * 64];  // per-wave P     (8 KB)

  const int lane = threadIdx.x & 31;
  const int wave = threadIdx.x >> 5;
  const int g    = lane >> 4;
  const int ln   = lane & 15;
  const int h    = blockIdx.y;
  const int q0   = (blockIdx.x * 4 + wave) * 16;

  const __bf16* qbase = qb   + (size_t)h * S_TOT * D_H;
  const __bf16* kbase = kbuf + (size_t)h * S_TOT * D_H;
  const __bf16* vbase = vt   + (size_t)h * D_H * S_TOT;

  // ---- Q fragments (A layout), head-dim split into two K=32 chunks
  FragB16 qa0, qa1;
  {
    const __bf16* qr = qbase + (size_t)(q0 + ln) * D_H;
    qa0.q[0] = reinterpret_cast<const uint4*>(qr +      g * 8)[0];
    qa0.q[1] = reinterpret_cast<const uint4*>(qr + 16 + g * 8)[0];
    qa1.q[0] = reinterpret_cast<const uint4*>(qr + 32 + g * 8)[0];
    qa1.q[1] = reinterpret_cast<const uint4*>(qr + 48 + g * 8)[0];
  }

  f32x8 accf[4];
  float mrow[8], lrow[8];
#pragma unroll
  for (int t = 0; t < 4; ++t) accf[t] = (f32x8){0.f,0.f,0.f,0.f,0.f,0.f,0.f,0.f};
#pragma unroll
  for (int r = 0; r < 8; ++r) { mrow[r] = -1e30f; lrow[r] = 0.f; }

  const f32x8 zero = {0.f, 0.f, 0.f, 0.f, 0.f, 0.f, 0.f, 0.f};

  // staging role: each thread owns one 32-element half-row of K and of V^T
  const int rowi = threadIdx.x >> 1;            // 0..63 (key for K, d for V^T)
  const int half = (threadIdx.x & 1) * 32;

  auto loadChunk = [&](int key0, uint4* rk, uint4* rv) {
    const uint4* ks = reinterpret_cast<const uint4*>(
        kbase + (size_t)(key0 + rowi) * D_H + half);
    const uint4* vs = reinterpret_cast<const uint4*>(
        vbase + (size_t)rowi * S_TOT + key0 + half);
#pragma unroll
    for (int i = 0; i < 4; ++i) { rk[i] = ks[i]; rv[i] = vs[i]; }
  };
  auto storeChunk = [&](int buf, const uint4* rk, const uint4* rv) {
    uint4* kd = reinterpret_cast<uint4*>(&lsK[buf][rowi * 64 + half]);
    uint4* vd = reinterpret_cast<uint4*>(&lsV[buf][rowi * 64 + half]);
#pragma unroll
    for (int i = 0; i < 4; ++i) { kd[i] = rk[i]; vd[i] = rv[i]; }
  };

  uint4 rk[4], rv[4];
  loadChunk(0, rk, rv);
  storeChunk(0, rk, rv);

  for (int j = 0; j < S_TOT / 64; ++j) {
    const int cur  = j & 1;
    const bool more = (j + 1) < (S_TOT / 64);
    if (more) loadChunk((j + 1) * 64, rk, rv);  // overlap with compute below
    __syncthreads();                            // lsK/lsV[cur] ready

    // ---- S = Q K^T : 4 key subtiles x 2 WMMA (scale pre-folded into q)
    f32x8 s[4];
#pragma unroll
    for (int sub = 0; sub < 4; ++sub) {
      FragB16 kf;
      const __bf16* krow = &lsK[cur][(sub * 16 + ln) * 64];
      kf.q[0] = *reinterpret_cast<const uint4*>(krow + g * 8);
      kf.q[1] = *reinterpret_cast<const uint4*>(krow + g * 8 + 16);
      s[sub] = __builtin_amdgcn_wmma_f32_16x16x32_bf16(
          false, qa0.v, false, kf.v, (short)0, zero, false, false);
      kf.q[0] = *reinterpret_cast<const uint4*>(krow + 32 + g * 8);
      kf.q[1] = *reinterpret_cast<const uint4*>(krow + 32 + g * 8 + 16);
      s[sub] = __builtin_amdgcn_wmma_f32_16x16x32_bf16(
          false, qa1.v, false, kf.v, (short)0, s[sub], false, false);
    }

    // ---- online softmax (row = r + 8*g; 16 row values live in one half)
    float alpha[8];
#pragma unroll
    for (int r = 0; r < 8; ++r) {
      float mx = fmaxf(fmaxf(s[0][r], s[1][r]), fmaxf(s[2][r], s[3][r]));
#pragma unroll
      for (int off = 1; off < 16; off <<= 1)
        mx = fmaxf(mx, __shfl_xor(mx, off, 32));
      const float mnew = fmaxf(mrow[r], mx);
      alpha[r] = __expf(mrow[r] - mnew);
      mrow[r] = mnew;
      float sum = 0.f;
#pragma unroll
      for (int sub = 0; sub < 4; ++sub) {
        const float e = __expf(s[sub][r] - mnew);
        s[sub][r] = e;
        sum += e;
      }
#pragma unroll
      for (int off = 1; off < 16; off <<= 1)
        sum += __shfl_xor(sum, off, 32);
      lrow[r] = lrow[r] * alpha[r] + sum;
    }
#pragma unroll
    for (int t = 0; t < 4; ++t)
#pragma unroll
      for (int r = 0; r < 8; ++r) accf[t][r] *= alpha[r];

    // ---- P: C-layout -> A-layout through wave-private LDS (in-order DS)
    __bf16* P = lsP[wave];
#pragma unroll
    for (int sub = 0; sub < 4; ++sub)
#pragma unroll
      for (int r = 0; r < 8; ++r)
        P[(r + 8 * g) * 64 + sub * 16 + ln] = (__bf16)s[sub][r];
    wait_dscnt0();
    FragB16 pa0, pa1;          // P 16x64 -> two A fragments (keys 0..31, 32..63)
    pa0.q[0] = *reinterpret_cast<const uint4*>(&P[ln * 64 +      g * 8]);
    pa0.q[1] = *reinterpret_cast<const uint4*>(&P[ln * 64 + 16 + g * 8]);
    pa1.q[0] = *reinterpret_cast<const uint4*>(&P[ln * 64 + 32 + g * 8]);
    pa1.q[1] = *reinterpret_cast<const uint4*>(&P[ln * 64 + 48 + g * 8]);

    // ---- O += P V : V^T in LDS [d][key] -> contiguous key runs
#pragma unroll
    for (int t = 0; t < 4; ++t) {
      const __bf16* vrow = &lsV[cur][(t * 16 + ln) * 64];
      FragB16 vf;
      vf.q[0] = *reinterpret_cast<const uint4*>(vrow + g * 8);
      vf.q[1] = *reinterpret_cast<const uint4*>(vrow + g * 8 + 16);
      accf[t] = __builtin_amdgcn_wmma_f32_16x16x32_bf16(
          false, pa0.v, false, vf.v, (short)0, accf[t], false, false);
      vf.q[0] = *reinterpret_cast<const uint4*>(vrow + 32 + g * 8);
      vf.q[1] = *reinterpret_cast<const uint4*>(vrow + 32 + g * 8 + 16);
      accf[t] = __builtin_amdgcn_wmma_f32_16x16x32_bf16(
          false, pa1.v, false, vf.v, (short)0, accf[t], false, false);
    }

    if (more) storeChunk(cur ^ 1, rk, rv);      // fill next buffer (safe: all
                                                // waves passed this iter's barrier)
  }

  // ---- normalize and store: out[s][h*64 + d] fp32
#pragma unroll
  for (int r = 0; r < 8; ++r) {
    const float inv = 1.0f / lrow[r];
    const int row = q0 + r + 8 * g;
#pragma unroll
    for (int t = 0; t < 4; ++t)
      out[(size_t)row * C_DIM + h * 64 + t * 16 + ln] = accf[t][r] * inv;
  }
}

// ---------------------------------------------------------------------------
extern "C" void kernel_launch(void* const* d_in, const int* in_sizes, int n_in,
                              void* d_out, int out_size, void* d_ws, size_t ws_size,
                              hipStream_t stream) {
  (void)in_sizes; (void)n_in; (void)out_size; (void)ws_size;
  const float* hid = (const float*)d_in[0];   // hidden_states    [3840][1024]
  const float* enc = (const float*)d_in[1];   // encoder_hidden   [256][1024]
  const float* rc  = (const float*)d_in[2];   // rope_cos [3840][64]
  const float* rs  = (const float*)d_in[3];   // rope_sin [3840][64]
  const float* Wq  = (const float*)d_in[4];
  const float* bq  = (const float*)d_in[5];
  const float* Wk  = (const float*)d_in[6];
  const float* bk  = (const float*)d_in[7];
  const float* Wv  = (const float*)d_in[8];
  const float* bv  = (const float*)d_in[9];
  float* out = (float*)d_out;

  // workspace layout (bytes):
  //   wt  : 3*1024*1024*2 = 6 MiB  @ 0
  //   xbf : 4096*1024*2   = 8 MiB  @ 6 MiB
  //   q   : 4096*1024*2   = 8 MiB  @ 14 MiB   [h][s][d]
  //   k   : 4096*1024*2   = 8 MiB  @ 22 MiB   [h][s][d]
  //   vt  : 4096*1024*2   = 8 MiB  @ 30 MiB   [h][d][s]
  char* ws = (char*)d_ws;
  __bf16* wt  = (__bf16*)(ws);
  __bf16* xbf = (__bf16*)(ws + (size_t)6  * 1024 * 1024);
  __bf16* qbw = (__bf16*)(ws + (size_t)14 * 1024 * 1024);
  __bf16* kbw = (__bf16*)(ws + (size_t)22 * 1024 * 1024);
  __bf16* vtw = (__bf16*)(ws + (size_t)30 * 1024 * 1024);

  xcat_bf16_kernel<<<dim3((S_TOT * C_DIM / 8) / 256), 256, 0, stream>>>(hid, enc, xbf);
  wt_transpose_kernel<<<dim3(C_DIM / 16, C_DIM / 16, 3), dim3(16, 16), 0, stream>>>(
      Wq, Wk, Wv, wt);
  qkv_rope_kernel<<<dim3(S_TOT / 16 / 4, C_DIM / 64, 3), 128, 0, stream>>>(
      xbf, wt, rc, rs, bq, bk, bv, qbw, kbw, vtw);
  attn_kernel<<<dim3(S_TOT / 64, H_N), 128, 0, stream>>>(qbw, kbw, vtw, out);
}
